// QuantLinear_24721831756571
// MI455X (gfx1250) — compile-verified, matrix operand order checked by reading, not run on hardware
//
#include <hip/hip_runtime.h>
#include <hip/hip_bf16.h>
#include <stdint.h>

typedef __attribute__((ext_vector_type(16))) _Float16 v16h;
typedef __attribute__((ext_vector_type(8)))  float    v8f;
typedef __attribute__((ext_vector_type(2)))  _Float16 h2;

#define OUT_F 4096
#define IN_F  11008
#define M_ROWS 64
#define NSPLIT 8
#define KSEG (IN_F / NSPLIT)        // 1376
#define KCHUNKS (KSEG / 32)         // 43
#define KB 4                        // chunks staged per barrier round (tail = 3)
#define WAVES_PER_WG 8
#define COLS_PER_WG (WAVES_PER_WG * 16)   // 128

union U32H2  { uint32_t u; h2 h; };
union BFrag  { v16h v; h2 h[8]; };

// ---------------------------------------------------------------------------
// Kernel 0: out[m, n] = bias[n]  (re-initializes poisoned d_out every launch)
// ---------------------------------------------------------------------------
__global__ void bias_init_kernel(const float* __restrict__ bias,
                                 float* __restrict__ out) {
    int idx = blockIdx.x * blockDim.x + threadIdx.x;
    if (idx < M_ROWS * OUT_F) {
        out[idx] = bias[idx & (OUT_F - 1)];
    }
}

// ---------------------------------------------------------------------------
// Kernel 1: convert x to f16 once into workspace (removes cvt from hot loop)
// ---------------------------------------------------------------------------
__global__ void x_to_f16_kernel(const float* __restrict__ x,
                                _Float16* __restrict__ x16) {
    int i = (blockIdx.x * blockDim.x + threadIdx.x) * 4;
    if (i < M_ROWS * IN_F) {
        #pragma unroll
        for (int t = 0; t < 4; ++t) x16[i + t] = (_Float16)x[i + t];
    }
}

// ---------------------------------------------------------------------------
// Kernel 2: dequant Q4_K-style weights on the fly (f16 magic-number path),
// WMMA f16->f32 accumulate, K split NSPLIT ways, atomic-add partials.
// Wave: 16 output columns x full M=64 (4 accumulators of 16x16).
// ---------------------------------------------------------------------------
__global__ __launch_bounds__(256)
void q4k_wmma_kernel(const _Float16* __restrict__ x16,
                     const int*   __restrict__ packed,
                     const float* __restrict__ dsc,
                     const float* __restrict__ dmn,
                     const int*   __restrict__ scales,
                     const int*   __restrict__ mins,
                     float*       __restrict__ out) {
    __shared__ _Float16 xs[KB * M_ROWS * 32];   // 16 KB: KB chunks of x[64][32]

    const int tid    = threadIdx.x;
    const int lane   = tid & 31;
    const int wave   = tid >> 5;
    const int col16  = lane & 15;          // column within the wave's 16-wide tile
    const int hiHalf = (lane >> 4) & 1;    // lanes 16-31 hold the upper K octet

    const int n     = blockIdx.x * COLS_PER_WG + wave * 16 + col16; // output column
    const int kbase = blockIdx.y * KSEG;

    v8f acc[4];
    #pragma unroll
    for (int mt = 0; mt < 4; ++mt) acc[mt] = (v8f){};

    for (int kc0 = 0; kc0 < KCHUNKS; kc0 += KB) {
        const int nb = (KCHUNKS - kc0 < KB) ? (KCHUNKS - kc0) : KB;

        // ---- stage x16[0:64, ...] into LDS: one b128 copy per thread/chunk ----
        for (int j = 0; j < nb; ++j) {
            const int kglob = kbase + (kc0 + j) * 32;
            const int e = tid * 8;             // 2048 halves per chunk
            const int m = e >> 5;
            const int k = e & 31;
            const uint4* src = (const uint4*)(x16 + m * IN_F + kglob + k);
            *(uint4*)(xs + j * (M_ROWS * 32) + e) = *src;
        }
        __syncthreads();

        for (int j = 0; j < nb; ++j) {
            const int kglob = kbase + (kc0 + j) * 32;

            // ---- per-lane scale/min: one sub-block covers this 32-K chunk ----
            const int   super = n * (IN_F / 256) + (kglob >> 8);
            const int   sub   = (kglob >> 5) & 7;
            const float dd    = dsc[super];
            const float mm    = dmn[super];
            const float sc    = (float)scales[super * 8 + sub];
            const float mn    = (float)mins[super * 8 + sub];
            const float wS    = dd * sc * (1.0f / (63.0f * 15.0f));
            const float wB    = dd * mn * (1.0f / 63.0f) + mm;
            // magic-number dequant: 0x6400|q is f16 (1024+q), so
            // w = q*wS + wB = (1024+q)*wS + (wB - 1024*wS)
            const _Float16 wSh = (_Float16)wS;
            const _Float16 wBh = (_Float16)(wB - 1024.0f * wS);
            const h2 wS2 = { wSh, wSh };
            const h2 wB2 = { wBh, wBh };

            // ---- dequantize B fragment: column n, k = kglob .. kglob+31 ----
            // B 16-bit layout: half-pair p holds consecutive k = kp, kp+1,
            // kp = (p/4)*16 + hiHalf*8 + 2*(p%4). One packed word = one pair.
            const int* pw = packed + ((n * IN_F + kglob) >> 1);
            __builtin_prefetch(pw + 16, 0, 1);     // next chunk's packed words
            BFrag bf;
            #pragma unroll
            for (int p = 0; p < 8; ++p) {
                const int kp = ((p >> 2) << 4) + (hiHalf << 3) + ((p & 3) << 1);
                const uint32_t w = (uint32_t)pw[kp >> 1];
                U32H2 t;
                t.u = 0x64006400u | (w & 15u) | ((w & 0xF0u) << 12);
                bf.h[p] = t.h * wS2 + wB2;         // v_pk_fma_f16
            }

            // ---- preload all 4 A fragments, then 4 back-to-back WMMAs ----
            v16h a[4];
            #pragma unroll
            for (int mt = 0; mt < 4; ++mt) {
                const _Float16* row =
                    xs + j * (M_ROWS * 32) + (mt * 16 + col16) * 32;
                #pragma unroll
                for (int p = 0; p < 8; ++p) {
                    const int k = ((p >> 2) << 4) + (hiHalf << 3) + ((p & 3) << 1);
                    a[mt][2 * p]     = row[k];
                    a[mt][2 * p + 1] = row[k + 1];
                }
            }
            #pragma unroll
            for (int mt = 0; mt < 4; ++mt) {
                acc[mt] = __builtin_amdgcn_wmma_f32_16x16x32_f16(
                    false, a[mt], false, bf.v, (short)0, acc[mt], false, false);
            }
        }
        __syncthreads();
    }

    // ---- C/D layout: lane holds column n, rows (hiHalf*8 + r) per M-tile ----
    #pragma unroll
    for (int mt = 0; mt < 4; ++mt) {
        #pragma unroll
        for (int r = 0; r < 8; ++r) {
            const int m = mt * 16 + hiHalf * 8 + r;
            atomicAdd(&out[m * OUT_F + n], acc[mt][r]);
        }
    }
}

// ---------------------------------------------------------------------------
extern "C" void kernel_launch(void* const* d_in, const int* in_sizes, int n_in,
                              void* d_out, int out_size, void* d_ws, size_t ws_size,
                              hipStream_t stream) {
    const float* x      = (const float*)d_in[0];
    const int*   packed = (const int*)  d_in[1];
    const float* d      = (const float*)d_in[2];
    const float* dmin   = (const float*)d_in[3];
    const int*   scales = (const int*)  d_in[4];
    const int*   mins   = (const int*)  d_in[5];
    const float* bias   = (const float*)d_in[6];
    float*       out    = (float*)d_out;
    _Float16*    x16    = (_Float16*)d_ws;     // 64*11008*2 = 1.38 MB scratch

    bias_init_kernel<<<(M_ROWS * OUT_F + 255) / 256, 256, 0, stream>>>(bias, out);
    x_to_f16_kernel<<<(M_ROWS * IN_F / 4 + 255) / 256, 256, 0, stream>>>(x, x16);

    dim3 grid(OUT_F / COLS_PER_WG, NSPLIT);   // 32 x 8 = 256 workgroups
    q4k_wmma_kernel<<<grid, 256, 0, stream>>>(x16, packed, d, dmin, scales, mins, out);
}
